// SelfAttention2D_10797547782878
// MI455X (gfx1250) — matrix operand order repaired and checked by reading, not compiled, not run on hardware
//
#include <hip/hip_runtime.h>

// ---------------- problem constants ----------------
#define R_DIM   128
#define C_DIM   512
#define E_DIM   256
#define HEADS   8
#define HD      32                       // head dim
#define M_ROWS  (R_DIM * C_DIM)          // 65536 flattened (r,c) rows
#define OUT_ELEMS ((size_t)M_ROWS * E_DIM)       // 16777216 -> probs offset
#define SCALING 0.015625f                // 32^-0.5 / sqrt(128) = 1/64

typedef __attribute__((ext_vector_type(16))) __bf16 v16bf;
typedef __attribute__((ext_vector_type(8)))  float  v8f;

// ---------------- WMMA fragment gathers (wave32; contiguous k order shared by A and B) ----------------
// A fragment: 16x32 (MxK) bf16. lanes 0-15: rows M=0..15, K 0..15;
// lanes 16-31: same rows, K 16..31. VGPR v packs K=kbase,kbase+1.
__device__ __forceinline__ v16bf frag_a_mk(const __bf16* lds, int ldm, int lane) {
    int m = lane & 15, half = lane >> 4;
    v16bf a;
#pragma unroll
    for (int v = 0; v < 8; ++v) {
        int kbase = half * 16 + 2 * v;
        a[2 * v]     = lds[m * ldm + kbase];
        a[2 * v + 1] = lds[m * ldm + kbase + 1];
    }
    return a;
}

// B fragment: 32x16 (KxN) bf16 from LDS stored [n][k].
// lane n = lane&15; lanes 0-15 hold K=0..15, lanes 16-31 hold K=16..31.
__device__ __forceinline__ v16bf frag_b_nk(const __bf16* lds, int ldn, int lane) {
    int n = lane & 15, half = lane >> 4;
    v16bf b;
#pragma unroll
    for (int v = 0; v < 8; ++v) {
        int k = half * 16 + 2 * v;
        b[2 * v]     = lds[n * ldn + k];
        b[2 * v + 1] = lds[n * ldn + k + 1];
    }
    return b;
}

// B fragment from LDS stored [k][n] (row k holds ldn columns).
__device__ __forceinline__ v16bf frag_b_kn(const __bf16* lds, int ldn, int lane) {
    int n = lane & 15, half = lane >> 4;
    v16bf b;
#pragma unroll
    for (int v = 0; v < 8; ++v) {
        int k = half * 16 + 2 * v;
        b[2 * v]     = lds[k * ldn + n];
        b[2 * v + 1] = lds[(k + 1) * ldn + n];
    }
    return b;
}

#define WMMA_BF16(a, b, c) \
    __builtin_amdgcn_wmma_f32_16x16x32_bf16(false, (a), false, (b), (short)0, (c), false, false)

// Async global->LDS 128-bit copy (CDNA5, tracked with ASYNCcnt).
__device__ __forceinline__ void async_b128(unsigned int lds_addr, const void* gsrc) {
    asm volatile("global_load_async_to_lds_b128 %0, %1, off"
                 :: "v"(lds_addr), "v"(gsrc) : "memory");
}
#define S_WAIT_ASYNCCNT(N) asm volatile("s_wait_asynccnt " #N ::: "memory")

// ---------------- kernel 1: convert weights f32 -> bf16 ----------------
// wsW rows 0-255: q_w, 256-511: k_w, 512-767: v_w, 768-1023: o_w (all [n][k] = original layout)
__global__ void convert_weights_k(const float* __restrict__ qw, const float* __restrict__ kw,
                                  const float* __restrict__ vw, const float* __restrict__ ow,
                                  __bf16* __restrict__ wsW) {
    int i = blockIdx.x * blockDim.x + threadIdx.x;   // 0 .. 262143
    int m = i >> 16, idx = i & 0xFFFF;
    const float* mats[4] = { qw, kw, vw, ow };
    wsW[i] = (__bf16)mats[m][idx];
}

// ---------------- kernel 2: fused QKV projection GEMM ----------------
// C(65536 x 768) = X(65536 x 256) * Wqkv^T ; epilogue: +bias, q-scale/mask, bf16 out.
__global__ void qkv_gemm_k(const float* __restrict__ x, const __bf16* __restrict__ wsW,
                           const float* __restrict__ qb, const float* __restrict__ kb,
                           const float* __restrict__ vb, const unsigned char* __restrict__ pmask,
                           __bf16* __restrict__ Q, __bf16* __restrict__ K, __bf16* __restrict__ V) {
    __shared__ __align__(16) __bf16 As[64][32];
    __shared__ __align__(16) __bf16 Bs[64][32];
    int tid = threadIdx.x, lane = tid & 31, wave = tid >> 5;
    int m0 = blockIdx.x * 64;
    int n0 = blockIdx.y * 64;          // 0..767
    v8f acc[4] = {};
    for (int ks = 0; ks < E_DIM; ks += 32) {
        // A tile: gather all f32 loads first, then convert -> LDS
        float4 fr[4];
#pragma unroll
        for (int t = 0; t < 4; ++t) {
            int idx = tid + t * 128;                 // 0..511
            int row = idx >> 3, c4 = (idx & 7) * 4;
            fr[t] = *(const float4*)(x + (size_t)(m0 + row) * E_DIM + ks + c4);
        }
        uint4 br[2];
#pragma unroll
        for (int t = 0; t < 2; ++t) {
            int idx = tid + t * 128;                 // 0..255
            int row = idx >> 2, q = idx & 3;
            br[t] = *(const uint4*)(wsW + (size_t)(n0 + row) * E_DIM + ks + q * 8);
        }
#pragma unroll
        for (int t = 0; t < 4; ++t) {
            int idx = tid + t * 128;
            int row = idx >> 3, c4 = (idx & 7) * 4;
            As[row][c4 + 0] = (__bf16)fr[t].x; As[row][c4 + 1] = (__bf16)fr[t].y;
            As[row][c4 + 2] = (__bf16)fr[t].z; As[row][c4 + 3] = (__bf16)fr[t].w;
        }
#pragma unroll
        for (int t = 0; t < 2; ++t) {
            int idx = tid + t * 128;
            int row = idx >> 2, q = idx & 3;
            *(uint4*)&Bs[row][q * 8] = br[t];
        }
        __syncthreads();
        v16bf a = frag_a_mk(&As[wave * 16][0], 32, lane);
#pragma unroll
        for (int nf = 0; nf < 4; ++nf) {
            v16bf b = frag_b_nk(&Bs[nf * 16][0], 32, lane);
            acc[nf] = WMMA_BF16(a, b, acc[nf]);
        }
        __syncthreads();
    }
    int half = lane >> 4, nl = lane & 15;
#pragma unroll
    for (int nf = 0; nf < 4; ++nf) {
        int ng = n0 + nf * 16 + nl;
        float bias = (ng < 256) ? qb[ng] : (ng < 512) ? kb[ng - 256] : vb[ng - 512];
#pragma unroll
        for (int v = 0; v < 8; ++v) {
            int mg = m0 + wave * 16 + half * 8 + v;          // = r*C + c (B==1)
            float val = acc[nf][v] + bias;
            if (ng < 256) {
                float keep = pmask[mg] ? 0.0f : 1.0f;
                Q[(size_t)mg * E_DIM + ng] = (__bf16)(val * SCALING * keep);
            } else if (ng < 512) {
                K[(size_t)mg * E_DIM + (ng - 256)] = (__bf16)val;
            } else {
                V[(size_t)mg * E_DIM + (ng - 512)] = (__bf16)val;
            }
        }
    }
}

// ---------------- kernel 3: attention scores (async double-buffered) ----------------
// scores[h,i,j] = sum_{r,d} Q[(r,i),h,d] * K[(r,j),h,d] ; K-dim = 128 r-slices of 32.
__global__ void attn_scores_k(const __bf16* __restrict__ Q, const __bf16* __restrict__ K,
                              const unsigned char* __restrict__ pmask, float* __restrict__ probs) {
    __shared__ __align__(16) __bf16 Qs[2][64][32];
    __shared__ __align__(16) __bf16 Ks[2][64][32];
    int tid = threadIdx.x, lane = tid & 31, wave = tid >> 5;
    int jt = blockIdx.x * 64, it = blockIdx.y * 64, h = blockIdx.z;

    // per-thread transfer coordinates (2 x b128 per tile per thread)
    int row0 = tid >> 2,        q0 = (tid & 3);
    int row1 = (tid + 128) >> 2, q1 = ((tid + 128) & 3);
    unsigned int qbase = (unsigned int)(size_t)&Qs[0][0][0];
    unsigned int kbase = (unsigned int)(size_t)&Ks[0][0][0];

    v8f acc[4] = {};
#define ISSUE_TILE(r, buf)                                                                   \
    do {                                                                                     \
        unsigned int qo = qbase + (unsigned int)((buf) * 4096 + row0 * 64 + q0 * 16);        \
        unsigned int ko = kbase + (unsigned int)((buf) * 4096 + row0 * 64 + q0 * 16);        \
        async_b128(qo, Q + ((size_t)(r) * C_DIM + it + row0) * E_DIM + h * HD + q0 * 8);     \
        async_b128(ko, K + ((size_t)(r) * C_DIM + jt + row0) * E_DIM + h * HD + q0 * 8);     \
        qo = qbase + (unsigned int)((buf) * 4096 + row1 * 64 + q1 * 16);                     \
        ko = kbase + (unsigned int)((buf) * 4096 + row1 * 64 + q1 * 16);                     \
        async_b128(qo, Q + ((size_t)(r) * C_DIM + it + row1) * E_DIM + h * HD + q1 * 8);     \
        async_b128(ko, K + ((size_t)(r) * C_DIM + jt + row1) * E_DIM + h * HD + q1 * 8);     \
    } while (0)

    ISSUE_TILE(0, 0);
    for (int r = 0; r < R_DIM; ++r) {
        int cur = r & 1;
        if (r + 1 < R_DIM) {
            ISSUE_TILE(r + 1, cur ^ 1);
            S_WAIT_ASYNCCNT(4);          // previous tile's 4 transfers done
        } else {
            S_WAIT_ASYNCCNT(0);
        }
        __syncthreads();
        v16bf a = frag_a_mk(&Qs[cur][wave * 16][0], 32, lane);
#pragma unroll
        for (int nf = 0; nf < 4; ++nf) {
            v16bf b = frag_b_nk(&Ks[cur][nf * 16][0], 32, lane);
            acc[nf] = WMMA_BF16(a, b, acc[nf]);
        }
        __syncthreads();                 // all waves done reading before buffer reuse
    }
#undef ISSUE_TILE

    int half = lane >> 4, nl = lane & 15;
#pragma unroll
    for (int nf = 0; nf < 4; ++nf) {
        int jg = jt + nf * 16 + nl;
        bool masked = pmask[jg] != 0;                        // col_mask = padding_mask[0,0,:]
#pragma unroll
        for (int v = 0; v < 8; ++v) {
            int ig = it + wave * 16 + half * 8 + v;
            float s = masked ? -10000.0f : acc[nf][v];
            probs[((size_t)h * C_DIM + ig) * C_DIM + jg] = s;
        }
    }
}

// ---------------- kernel 4: softmax over j (in place) + bf16 copy ----------------
__global__ void softmax_k(float* __restrict__ probs, __bf16* __restrict__ Pb) {
    __shared__ float red[256];
    size_t row = blockIdx.x;                // h*512 + i
    float* p = probs + row * C_DIM;
    int tid = threadIdx.x;
    float a = p[tid], b = p[tid + 256];
    red[tid] = fmaxf(a, b);
    __syncthreads();
    for (int s = 128; s > 0; s >>= 1) {
        if (tid < s) red[tid] = fmaxf(red[tid], red[tid + s]);
        __syncthreads();
    }
    float mx = red[0];
    __syncthreads();
    float ea = __expf(a - mx), eb = __expf(b - mx);
    red[tid] = ea + eb;
    __syncthreads();
    for (int s = 128; s > 0; s >>= 1) {
        if (tid < s) red[tid] += red[tid + s];
        __syncthreads();
    }
    float inv = 1.0f / red[0];
    float pa = ea * inv, pb = eb * inv;
    p[tid] = pa;       p[tid + 256] = pb;
    Pb[row * C_DIM + tid] = (__bf16)pa;
    Pb[row * C_DIM + tid + 256] = (__bf16)pb;
}

// ---------------- kernel 5: context GEMM ----------------
// ctx[(r,i),h,d] = sum_j P[h,i,j] * V[(r,j),h,d] ; per head M=512(i), N=4096(r,d), K=512(j)
__global__ void context_gemm_k(const __bf16* __restrict__ Pb, const __bf16* __restrict__ V,
                               __bf16* __restrict__ Ctx) {
    __shared__ __align__(16) __bf16 Ps[64][32];
    __shared__ __align__(16) __bf16 Vs[32][64];      // [j'][n] with n -> (r,d)
    int tid = threadIdx.x, lane = tid & 31, wave = tid >> 5;
    int nt = blockIdx.x;                 // 64 tiles of 64 over N=4096
    int it = blockIdx.y * 64;
    int h  = blockIdx.z;
    int r0 = nt * 2;
    v8f acc[4] = {};
    for (int js = 0; js < C_DIM; js += 32) {
        uint4 pr[2], vr[2];
#pragma unroll
        for (int t = 0; t < 2; ++t) {
            int idx = tid + t * 128;                 // Ps: 0..255
            int row = idx >> 2, q = idx & 3;
            pr[t] = *(const uint4*)(Pb + ((size_t)h * C_DIM + it + row) * C_DIM + js + q * 8);
            int jp = idx >> 3, seg = idx & 7;        // Vs: seg = (r half)<<2 | d-quarter
            int rr = r0 + (seg >> 2), d0 = (seg & 3) * 8;
            vr[t] = *(const uint4*)(V + ((size_t)rr * C_DIM + js + jp) * E_DIM + h * HD + d0);
        }
#pragma unroll
        for (int t = 0; t < 2; ++t) {
            int idx = tid + t * 128;
            int row = idx >> 2, q = idx & 3;
            *(uint4*)&Ps[row][q * 8] = pr[t];
            int jp = idx >> 3, seg = idx & 7;
            *(uint4*)&Vs[jp][(seg >> 2) * 32 + (seg & 3) * 8] = vr[t];
        }
        __syncthreads();
        v16bf a = frag_a_mk(&Ps[wave * 16][0], 32, lane);
#pragma unroll
        for (int nf = 0; nf < 4; ++nf) {
            v16bf b = frag_b_kn(&Vs[0][nf * 16], 64, lane);
            acc[nf] = WMMA_BF16(a, b, acc[nf]);
        }
        __syncthreads();
    }
    int half = lane >> 4, nl = lane & 15;
#pragma unroll
    for (int nf = 0; nf < 4; ++nf) {
        int ncol = nt * 64 + nf * 16 + nl;
        int rr = ncol >> 5, d = ncol & 31;
#pragma unroll
        for (int v = 0; v < 8; ++v) {
            int ig = it + wave * 16 + half * 8 + v;
            Ctx[((size_t)rr * C_DIM + ig) * E_DIM + h * HD + d] = (__bf16)acc[nf][v];
        }
    }
}

// ---------------- kernel 6: output projection ----------------
__global__ void out_gemm_k(const __bf16* __restrict__ Ctx, const __bf16* __restrict__ Wo,
                           const float* __restrict__ ob, float* __restrict__ out) {
    __shared__ __align__(16) __bf16 As[64][32];
    __shared__ __align__(16) __bf16 Bs[64][32];
    int tid = threadIdx.x, lane = tid & 31, wave = tid >> 5;
    int m0 = blockIdx.x * 64, n0 = blockIdx.y * 64;
    v8f acc[4] = {};
    for (int ks = 0; ks < E_DIM; ks += 32) {
        uint4 ar[2], br[2];
#pragma unroll
        for (int t = 0; t < 2; ++t) {
            int idx = tid + t * 128;
            int row = idx >> 2, q = idx & 3;
            ar[t] = *(const uint4*)(Ctx + (size_t)(m0 + row) * E_DIM + ks + q * 8);
            br[t] = *(const uint4*)(Wo + (size_t)(n0 + row) * E_DIM + ks + q * 8);
        }
#pragma unroll
        for (int t = 0; t < 2; ++t) {
            int idx = tid + t * 128;
            int row = idx >> 2, q = idx & 3;
            *(uint4*)&As[row][q * 8] = ar[t];
            *(uint4*)&Bs[row][q * 8] = br[t];
        }
        __syncthreads();
        v16bf a = frag_a_mk(&As[wave * 16][0], 32, lane);
#pragma unroll
        for (int nf = 0; nf < 4; ++nf) {
            v16bf b = frag_b_nk(&Bs[nf * 16][0], 32, lane);
            acc[nf] = WMMA_BF16(a, b, acc[nf]);
        }
        __syncthreads();
    }
    int half = lane >> 4, nl = lane & 15;
#pragma unroll
    for (int nf = 0; nf < 4; ++nf) {
        int ng = n0 + nf * 16 + nl;
        float bias = ob[ng];
#pragma unroll
        for (int v = 0; v < 8; ++v) {
            int mg = m0 + wave * 16 + half * 8 + v;
            out[(size_t)mg * E_DIM + ng] = acc[nf][v] + bias;
        }
    }
}

// ---------------- launcher ----------------
extern "C" void kernel_launch(void* const* d_in, const int* in_sizes, int n_in,
                              void* d_out, int out_size, void* d_ws, size_t ws_size,
                              hipStream_t stream) {
    const float*         x     = (const float*)d_in[0];
    const unsigned char* pmask = (const unsigned char*)d_in[1];
    const float* qw = (const float*)d_in[2];
    const float* qb = (const float*)d_in[3];
    const float* kw = (const float*)d_in[4];
    const float* kb = (const float*)d_in[5];
    const float* vw = (const float*)d_in[6];
    const float* vb = (const float*)d_in[7];
    const float* ow = (const float*)d_in[8];
    const float* ob = (const float*)d_in[9];

    float* out   = (float*)d_out;
    float* probs = out + OUT_ELEMS;              // attn_probs region of d_out

    // workspace layout (bytes):
    //   [0, 512K)      : bf16 weights (q,k,v,o stacked, 1024x256)
    //   [512K, +32M)   : Q bf16  -> reused as Ctx bf16
    //   [.. , +32M)    : K bf16  -> reused as probs bf16
    //   [.. , +32M)    : V bf16
    char* ws = (char*)d_ws;
    __bf16* wsW = (__bf16*)ws;
    __bf16* Q   = (__bf16*)(ws + 524288);
    __bf16* K   = (__bf16*)(ws + 524288 + 33554432ull);
    __bf16* V   = (__bf16*)(ws + 524288 + 2ull * 33554432ull);
    __bf16* Pb  = K;     // K dead after scores
    __bf16* Ctx = Q;     // Q dead after scores

    convert_weights_k<<<dim3(1024), dim3(256), 0, stream>>>(qw, kw, vw, ow, wsW);
    qkv_gemm_k<<<dim3(1024, 12), dim3(128), 0, stream>>>(x, wsW, qb, kb, vb, pmask, Q, K, V);
    attn_scores_k<<<dim3(8, 8, 8), dim3(128), 0, stream>>>(Q, K, pmask, probs);
    softmax_k<<<dim3(HEADS * C_DIM), dim3(256), 0, stream>>>(probs, Pb);
    context_gemm_k<<<dim3(64, 8, 8), dim3(128), 0, stream>>>(Pb, V, Ctx);
    out_gemm_k<<<dim3(1024, 4), dim3(128), 0, stream>>>(Ctx, wsW + 768 * 256, ob, out);
}
